// LocationSensitiveAttention_25975962206233
// MI455X (gfx1250) — matrix operand order repaired and checked by reading, not compiled, not run on hardware
//
#include <hip/hip_runtime.h>
#include <math.h>

// ---------------------------------------------------------------- constants
#define BATCH 32
#define TT 2048
#define DD 1024
#define KW 31          // conv kernel width

#define NKB   (DD / 32)      // 32 K-blocks of 32 for the big GEMM
#define NET   (DD / 16)      // 64 e-tiles
#define TM    32             // t rows per workgroup (2 m-tiles)
#define THREADS 512          // 16 waves
#define ETW   4              // e-tiles per wave (16 waves * 4 = 64)

// workspace layout (bytes)
#define WS_WMFRAG   0u                         // 32kb*64et*1024B = 2 MB
#define WS_WMFRAG_SZ (NKB * NET * 1024u)
#define WS_LOCFRAG  (WS_WMFRAG + WS_WMFRAG_SZ) // 2ukb*64et*1024B = 128 KB
#define WS_LOCFRAG_SZ (2u * NET * 1024u)
#define WS_Q        (WS_LOCFRAG + WS_LOCFRAG_SZ)   // 32*1024 f32 = 128 KB
#define WS_Q_SZ     (BATCH * DD * 4u)
#define WS_E        (WS_Q + WS_Q_SZ)               // 32*2048 f32 = 256 KB

// dynamic LDS layout for the main kernel (bytes)
#define L_AFRAG 0u                     // 32kb * 2mt * 1024B = 64 KB
#define L_UFRAG (NKB * 2u * 1024u)     // 2ukb * 2mt * 1024B = 4 KB
#define L_ESH   (L_UFRAG + 2u * 2u * 1024u)
#define L_TOTAL (L_ESH + TM * 4u)      // + 32 floats

typedef __attribute__((ext_vector_type(16))) __bf16 v16bf;
typedef __attribute__((ext_vector_type(8)))  float  v8f;

__device__ __forceinline__ unsigned int pack_bf2(float a, float b) {
    union { float f; unsigned int u; } ua, ub;
    ua.f = a; ub.f = b;
    unsigned int x = ua.u, y = ub.u;
    x = (x + 0x7FFFu + ((x >> 16) & 1u)) >> 16;   // round-to-nearest-even
    y = (y + 0x7FFFu + ((y >> 16) & 1u)) >> 16;
    return (x & 0xFFFFu) | (y << 16);
}

// ------------------------------------------------------------------ prep: Wm
// B-fragment layout (16-bit B, K=32): lane l holds column n=l&15,
// K half = (l<16 ? 0..15 : 16..31), dword j = K pair (2j,2j+1).
// ws dword chunk per (kb, et): 32 lanes * 32 B, fully coalesced on read.
__global__ void prep_wm(const float* __restrict__ Wm, unsigned int* __restrict__ frag) {
    int g = blockIdx.x * blockDim.x + threadIdx.x;       // 32*64*32 threads
    int kb = g >> 11;            // /(64*32)
    int et = (g >> 5) & 63;
    int lane = g & 31;
    int n = lane & 15;
    int e = et * 16 + n;
    int khalf = (lane < 16) ? 0 : 16;
    unsigned int out[8];
#pragma unroll
    for (int j = 0; j < 8; ++j) {
        int k = kb * 32 + khalf + 2 * j;
        const float* p = Wm + (size_t)e * DD + k;        // B[k][n] = Wm[e][k]
        out[j] = pack_bf2(p[0], p[1]);
    }
    unsigned int* dst = frag + ((size_t)(kb * NET + et) * 256) + lane * 8;
    *(uint4*)(dst)     = make_uint4(out[0], out[1], out[2], out[3]);
    *(uint4*)(dst + 4) = make_uint4(out[4], out[5], out[6], out[7]);
}

// -------------------------------------------------------------- prep: Wloc
// im2col weight W2[c,e], c = 2k+ch (c>=62 -> 0), same B-fragment layout.
__global__ void prep_wloc(const float* __restrict__ Wloc, unsigned int* __restrict__ frag) {
    int g = blockIdx.x * blockDim.x + threadIdx.x;       // 2*64*32 threads
    int ukb = g >> 11;
    int et = (g >> 5) & 63;
    int lane = g & 31;
    int n = lane & 15;
    int e = et * 16 + n;
    int khalf = (lane < 16) ? 0 : 16;
    unsigned int out[8];
#pragma unroll
    for (int j = 0; j < 8; ++j) {
        int c0 = ukb * 32 + khalf + 2 * j;
        float a = 0.f, b = 0.f;
        if (c0 < 2 * KW)     a = Wloc[((size_t)e * 2 + (c0 & 1)) * KW + (c0 >> 1)];
        if (c0 + 1 < 2 * KW) b = Wloc[((size_t)e * 2 + ((c0 + 1) & 1)) * KW + ((c0 + 1) >> 1)];
        out[j] = pack_bf2(a, b);
    }
    unsigned int* dst = frag + ((size_t)(ukb * NET + et) * 256) + lane * 8;
    *(uint4*)(dst)     = make_uint4(out[0], out[1], out[2], out[3]);
    *(uint4*)(dst + 4) = make_uint4(out[4], out[5], out[6], out[7]);
}

// ----------------------------------------------------------------- prep: q
__global__ void prep_q(const float* __restrict__ query, const float* __restrict__ Wq,
                       float* __restrict__ q_ws) {
    __shared__ float qsh[DD];
    int b = blockIdx.x;
    for (int i = threadIdx.x; i < DD; i += blockDim.x)
        qsh[i] = query[(size_t)b * DD + i];
    __syncthreads();
#pragma unroll
    for (int eo = 0; eo < 4; ++eo) {
        int e = threadIdx.x + eo * 256;
        const float4* wr = (const float4*)(Wq + (size_t)e * DD);
        const float4* qr = (const float4*)qsh;
        float acc = 0.f;
        for (int d4 = 0; d4 < DD / 4; ++d4) {
            float4 w = wr[d4], q = qr[d4];
            acc += w.x * q.x + w.y * q.y + w.z * q.z + w.w * q.w;
        }
        q_ws[(size_t)b * DD + e] = acc;
    }
}

// ------------------------------------------------------------ main WMMA kernel
// one WG = (b, 32 t-rows) x all 1024 e; emits e[b,t] = v . tanh(q + m + loc)
__global__ void __launch_bounds__(THREADS, 1)
attn_main(const float* __restrict__ memory,
          const float* __restrict__ prev_attn,
          const float* __restrict__ cum_attn,
          const float* __restrict__ vvec,
          const unsigned int* __restrict__ wmfrag,
          const unsigned int* __restrict__ locfrag,
          const float* __restrict__ q_ws,
          float* __restrict__ e_ws) {
    extern __shared__ char sm[];
    const int tid  = threadIdx.x;
    const int b    = blockIdx.x >> 6;            // / (TT/TM)
    const int t0   = (blockIdx.x & 63) * TM;

    if (tid < TM) *(float*)(sm + L_ESH + tid * 4) = 0.f;

    // ---- phase 1: load memory tile (coalesced f32x4), cvt->bf16, scatter to
    // A-fragment layout in LDS (16-bit A 16x32, contiguous k = 16*(lane/16) + elem, same order as B frags).
    {
        const float4* src = (const float4*)(memory + ((size_t)b * TT + t0) * DD);
#pragma unroll 4
        for (int it = 0; it < 16; ++it) {
            int idx4 = tid + it * THREADS;           // float4 index in 32x1024 tile
            float4 v = src[idx4];
            int off  = idx4 * 4;
            int r    = off >> 10;                    // t row 0..31
            int cg   = off & 1023;                   // column 0..1023
            int kb   = cg >> 5;
            int c    = cg & 31;
            int mt   = r >> 4;
            int lane = (r & 15) + ((c >> 4) << 4);
            int j    = (c & 15) >> 1;
            unsigned int d0 = pack_bf2(v.x, v.y);
            unsigned int d1 = pack_bf2(v.z, v.w);
            char* dst = sm + L_AFRAG + (size_t)(kb * 2 + mt) * 1024 + lane * 32 + j * 4;
            *(uint2*)dst = make_uint2(d0, d1);       // ds_store_b64
        }
    }

    // ---- phase 1b: build im2col U tile (32t x 64c) fragments (conv inputs)
    {
#pragma unroll
        for (int ii = 0; ii < 2; ++ii) {
            int id   = tid + ii * THREADS;           // 1024 frag dwords total
            int j    = id & 7;
            int lane = (id >> 3) & 31;
            int mt   = (id >> 8) & 1;
            int ukb  = (id >> 9) & 1;
            int m    = mt * 16 + (lane & 15);
            int khalf = (lane < 16) ? 0 : 16;
            int c0   = ukb * 32 + khalf + 2 * j;
            float a = 0.f, bb = 0.f;
#pragma unroll
            for (int p = 0; p < 2; ++p) {
                int c = c0 + p;
                float val = 0.f;
                if (c < 2 * KW) {
                    int k  = c >> 1;
                    int ts = t0 + m + k - (KW / 2);
                    if (ts >= 0 && ts < TT) {
                        const float* srcp = (c & 1) ? cum_attn : prev_attn;
                        val = srcp[(size_t)b * TT + ts];
                    }
                }
                if (p == 0) a = val; else bb = val;
            }
            char* dst = sm + L_UFRAG + (size_t)(ukb * 2 + mt) * 1024 + lane * 32 + j * 4;
            *(unsigned int*)dst = pack_bf2(a, bb);
        }
    }
    __syncthreads();

    // ---- phase 2: WMMA. wave w owns e-tiles [w*4, w*4+4), both m-tiles.
    const int w    = tid >> 5;
    const int lane = tid & 31;
    const int et0  = w * ETW;

    v8f acc[2][ETW];
#pragma unroll
    for (int mt = 0; mt < 2; ++mt)
#pragma unroll
        for (int et = 0; et < ETW; ++et)
            acc[mt][et] = (v8f){};

    for (int kb = 0; kb < NKB; ++kb) {
        v16bf af[2];
#pragma unroll
        for (int mt = 0; mt < 2; ++mt)
            af[mt] = *(const v16bf*)(sm + L_AFRAG + (size_t)(kb * 2 + mt) * 1024 + lane * 32);
        const char* bbase = (const char*)wmfrag + (size_t)(kb * NET + et0) * 1024;
        if (kb + 1 < NKB)
            __builtin_prefetch((const char*)wmfrag + (size_t)((kb + 1) * NET + et0) * 1024
                               + lane * 32, 0, 1);   // -> global_prefetch_b8
#pragma unroll
        for (int et = 0; et < ETW; ++et) {
            v16bf bf = *(const v16bf*)(bbase + (size_t)et * 1024 + lane * 32);
            acc[0][et] = __builtin_amdgcn_wmma_f32_16x16x32_bf16(
                false, af[0], false, bf, (short)0, acc[0][et], false, false);
            acc[1][et] = __builtin_amdgcn_wmma_f32_16x16x32_bf16(
                false, af[1], false, bf, (short)0, acc[1][et], false, false);
        }
    }
    // conv contribution: 2 extra K-steps from U-fragments
#pragma unroll
    for (int ukb = 0; ukb < 2; ++ukb) {
        v16bf af[2];
#pragma unroll
        for (int mt = 0; mt < 2; ++mt)
            af[mt] = *(const v16bf*)(sm + L_UFRAG + (size_t)(ukb * 2 + mt) * 1024 + lane * 32);
        const char* bbase = (const char*)locfrag + (size_t)(ukb * NET + et0) * 1024;
#pragma unroll
        for (int et = 0; et < ETW; ++et) {
            v16bf bf = *(const v16bf*)(bbase + (size_t)et * 1024 + lane * 32);
            acc[0][et] = __builtin_amdgcn_wmma_f32_16x16x32_bf16(
                false, af[0], false, bf, (short)0, acc[0][et], false, false);
            acc[1][et] = __builtin_amdgcn_wmma_f32_16x16x32_bf16(
                false, af[1], false, bf, (short)0, acc[1][et], false, false);
        }
    }

    // ---- phase 3: epilogue. acc element (mt,et)[r] is (t = t0+mt*16+r+(lane>=16?8:0),
    // e = (et0+et)*16 + (lane&15)).  e_out = sum_e tanh(acc+q[e]) * v[e].
    const int n = lane & 15;
    float qv[ETW], vv[ETW];
#pragma unroll
    for (int et = 0; et < ETW; ++et) {
        int e = (et0 + et) * 16 + n;
        qv[et] = q_ws[(size_t)b * DD + e];
        vv[et] = vvec[e];
    }
#pragma unroll
    for (int mt = 0; mt < 2; ++mt) {
#pragma unroll
        for (int r = 0; r < 8; ++r) {
            float s = 0.f;
#pragma unroll
            for (int et = 0; et < ETW; ++et)
                s += tanhf(acc[mt][et][r] + qv[et]) * vv[et];
            // reduce across the 16-lane half (xor bits 0..3 stay in-half)
            for (int off = 8; off >= 1; off >>= 1)
                s += __shfl_xor(s, off, 32);
            int t_local = mt * 16 + r + ((lane & 16) ? 8 : 0);
            if ((lane & 15) == 0)
                atomicAdd((float*)(sm + L_ESH) + t_local, s);   // ds_add_f32
        }
    }
    __syncthreads();
    if (tid < TM)
        e_ws[(size_t)b * TT + t0 + tid] = *(float*)(sm + L_ESH + tid * 4);
}

// --------------------------------------------------------------- softmax
__global__ void softmax_k(const float* __restrict__ e_ws, const int* __restrict__ mask,
                          float* __restrict__ a_out) {
    __shared__ float red[256];
    int b = blockIdx.x, tid = threadIdx.x;
    float vals[TT / 256];
    float mx = -INFINITY;
#pragma unroll
    for (int i = 0; i < TT / 256; ++i) {
        int t = tid + i * 256;
        float e = e_ws[(size_t)b * TT + t];
        if (mask[(size_t)b * TT + t]) e = -INFINITY;   // where(mask, -inf, e)
        vals[i] = e;
        mx = fmaxf(mx, e);
    }
    red[tid] = mx; __syncthreads();
    for (int s = 128; s > 0; s >>= 1) {
        if (tid < s) red[tid] = fmaxf(red[tid], red[tid + s]);
        __syncthreads();
    }
    mx = red[0]; __syncthreads();
    float sum = 0.f;
#pragma unroll
    for (int i = 0; i < TT / 256; ++i) {
        vals[i] = __expf(vals[i] - mx);
        sum += vals[i];
    }
    red[tid] = sum; __syncthreads();
    for (int s = 128; s > 0; s >>= 1) {
        if (tid < s) red[tid] += red[tid + s];
        __syncthreads();
    }
    float inv = 1.f / red[0];
#pragma unroll
    for (int i = 0; i < TT / 256; ++i)
        a_out[(size_t)b * TT + tid + i * 256] = vals[i] * inv;
}

// ------------------------------------------------------------------- ctx
// ctx[b,d] = sum_t a[b,t] * memory[b,t,d]   (GEMV, bandwidth bound)
__global__ void ctx_k(const float* __restrict__ a, const float* __restrict__ memory,
                      float* __restrict__ ctx) {
    __shared__ float cred[256];
    int blk = blockIdx.x;
    int b  = blk >> 4;               // 16 d-chunks of 64
    int dc = blk & 15;
    int dpos = threadIdx.x & 63;
    int tg   = threadIdx.x >> 6;
    int d = dc * 64 + dpos;
    float acc = 0.f;
    for (int t = tg; t < TT; t += 4)
        acc += a[(size_t)b * TT + t] * memory[((size_t)b * TT + t) * DD + d];
    cred[threadIdx.x] = acc;
    __syncthreads();
    if (tg == 0)
        ctx[(size_t)b * DD + d] = cred[dpos] + cred[64 + dpos] + cred[128 + dpos] + cred[192 + dpos];
}

// ---------------------------------------------------------------- launcher
extern "C" void kernel_launch(void* const* d_in, const int* in_sizes, int n_in,
                              void* d_out, int out_size, void* d_ws, size_t ws_size,
                              hipStream_t stream) {
    const float* query  = (const float*)d_in[0];
    const float* memory = (const float*)d_in[1];
    const float* prev   = (const float*)d_in[2];
    const float* cum    = (const float*)d_in[3];
    const int*   mask   = (const int*)  d_in[4];
    const float* Wq     = (const float*)d_in[5];
    const float* Wm     = (const float*)d_in[6];
    const float* Wloc   = (const float*)d_in[7];
    const float* v      = (const float*)d_in[8];

    char* ws = (char*)d_ws;
    unsigned int* wmfrag  = (unsigned int*)(ws + WS_WMFRAG);
    unsigned int* locfrag = (unsigned int*)(ws + WS_LOCFRAG);
    float*        q_ws    = (float*)(ws + WS_Q);
    float*        e_ws    = (float*)(ws + WS_E);

    float* ctx_out = (float*)d_out;               // [B, D]
    float* a_out   = (float*)d_out + BATCH * DD;  // [B, T]

    prep_wm  <<<256, 256, 0, stream>>>(Wm, wmfrag);
    prep_wloc<<<16,  256, 0, stream>>>(Wloc, locfrag);
    prep_q   <<<BATCH, 256, 0, stream>>>(query, Wq, q_ws);
    attn_main<<<BATCH * (TT / TM), THREADS, L_TOTAL, stream>>>(
        memory, prev, cum, v, wmfrag, locfrag, q_ws, e_ws);
    softmax_k<<<BATCH, 256, 0, stream>>>(e_ws, mask, a_out);
    ctx_k    <<<BATCH * 16, 256, 0, stream>>>(a_out, memory, ctx_out);
}
